// AttentionReceiver_14671608283703
// MI455X (gfx1250) — matrix operand order repaired and checked by reading, not compiled, not run on hardware
//
#include <hip/hip_runtime.h>
#include <hip/hip_bf16.h>

// Problem constants (from reference)
#define BSZ 1024
#define Lq  128
#define Ed  256
#define Hd  512
#define Td  20
// 4H = 2048, E+H = 768

typedef __bf16 v16bf __attribute__((ext_vector_type(16)));
typedef __bf16 v8bf  __attribute__((ext_vector_type(8)));
typedef float  v8f   __attribute__((ext_vector_type(8)));

__device__ __forceinline__ __bf16 f2bf(float f) {
  unsigned u = __float_as_uint(f);
  unsigned r = (u + 0x7FFFu + ((u >> 16) & 1u)) >> 16;   // round-to-nearest-even
  unsigned short s = (unsigned short)r;
  return __builtin_bit_cast(__bf16, s);
}

// ---- Packed A-fragment layout (ISA 7.12.2, 16-bit A 16x32 per K-tile) ----
// Tile kt holds K = kt*32..kt*32+31. Within a tile: lane = hi*16 + m where
// hi = (ko>>3)&1 (lanes 0-15: ko in {0..7,16..23}; lanes 16-31: {8..15,24..31});
// element e = (ko&7) + ((ko>>4)<<3).  Storage: A[(kt*32 + lane)*16 + e].

// consumer: one 32B contiguous LDS vector load per fragment
__device__ __forceinline__ v16bf load_A_packed(const __bf16* A, int kt, int lane) {
  return *(const v16bf*)(A + ((kt * 32) + lane) * 16);
}
// producer: scalar scatter (k fixed per thread, m varies -> stride 16 elems)
__device__ __forceinline__ void a_store(__bf16* A, int m, int k, float v) {
  int kt = k >> 5, ko = k & 31;
  int hi = (ko >> 3) & 1;
  int e  = (ko & 7) + ((ko >> 4) << 3);
  A[((kt * 32) + hi * 16 + m) * 16 + e] = f2bf(v);
}
// producer: pointer to an 8-wide chunk (k aligned to 8) for vector stores
__device__ __forceinline__ __bf16* a_ptr8(__bf16* A, int m, int k) {
  int kt = k >> 5, ko = k & 31;
  int hi = (ko >> 3) & 1;
  int e  = (ko >> 4) << 3;
  return A + ((kt * 32) + hi * 16 + m) * 16 + e;
}

// B-fragment (32x16 bf16): B[k,n] = W[n0+n, k0+k]; lane n = lane&15, hi = lane>>4
// -> each lane reads k = 8*hi..+7 and 16+8*hi..+7 (two 16B chunks) from one W row.
__device__ __forceinline__ v16bf load_B(const __bf16* W, int ldk, int n0, int k0, int lane) {
  const __bf16* p = W + (size_t)(n0 + (lane & 15)) * ldk + k0 + ((lane >> 4) << 3);
  v8bf lo = *(const v8bf*)p;
  v8bf hi = *(const v8bf*)(p + 16);
  return __builtin_shufflevector(lo, hi, 0, 1, 2, 3, 4, 5, 6, 7, 8, 9, 10, 11, 12, 13, 14, 15);
}

#define WMMA_BF16(a, b, c) \
  __builtin_amdgcn_wmma_f32_16x16x32_bf16(false, (a), false, (b), (short)0, (c), false, false)

__device__ __forceinline__ float sigf(float x) { return 1.0f / (1.0f + __expf(-x)); }

__global__ void convert_f32_to_bf16(const float* __restrict__ src,
                                    __bf16* __restrict__ dst, int n) {
  int i = blockIdx.x * blockDim.x + threadIdx.x;
  if (i < n) dst[i] = f2bf(src[i]);
}

__global__ void zero_tail(float* __restrict__ out, int base, int n) {
  int i = blockIdx.x * blockDim.x + threadIdx.x;
  if (i < n) out[base + i] = 0.0f;
}

// Persistent recurrent kernel: one WG = 16 batch rows, 16 waves (512 thr, wave32).
__global__ __launch_bounds__(512) void attn_lstm_kernel(
    const int* __restrict__ msg, const float* __restrict__ M,
    const int* __restrict__ mask, const int* __restrict__ lengths,
    const float* __restrict__ emb,
    const float* __restrict__ bq, const float* __restrict__ bc,
    const float* __restrict__ b_ih, const float* __restrict__ b_hh,
    const __bf16* __restrict__ Wq_bf, const __bf16* __restrict__ Wc_bf,
    const __bf16* __restrict__ Wih_bf, const __bf16* __restrict__ Whh_bf,
    float* __restrict__ out) {
  extern __shared__ char smem[];
  float*  sh_c    = (float*)smem;                 // 16*512 f32   32KB
  float*  sh_pq   = sh_c + 16 * Hd;               // 16*256 f32   16KB
  float*  sh_w    = sh_pq + 16 * Ed;              // 16*128 f32    8KB
  __bf16* sh_hA   = (__bf16*)(sh_w + 16 * Lq);    // 16x512 bf16  16KB (A-packed)
  __bf16* sh_attA = sh_hA   + 16 * Hd;            // 16x512 bf16  16KB (A-packed)
  __bf16* sh_appA = sh_attA + 16 * Hd;            // 16x256 bf16   8KB (A-packed)
  __bf16* sh_xA   = sh_appA + 16 * Ed;            // 16x256 bf16   8KB (A-packed)
  // total 104KB -> 3 WGs per 320KB WGP

  const int tid  = threadIdx.x;
  const int lane = tid & 31;
  const int wave = tid >> 5;            // 0..15
  const int b0   = blockIdx.x * 16;

  for (int i = tid; i < 16 * Hd; i += 512) sh_c[i] = 0.0f;
  {
    unsigned* z = (unsigned*)sh_hA;     // zero bf16 h (16*512 elems = 4096 dwords)
    for (int i = tid; i < 16 * Hd / 2; i += 512) z[i] = 0u;
  }
  __syncthreads();

  int row_len = lengths[b0 + wave];
  row_len = row_len < 0 ? 0 : (row_len > Td ? Td : row_len);

  for (int t = 0; t <= Td; ++t) {
    // ---- GEMM1: pq = h @ Wq^T + bq  (16x256, K=512); wave owns one N-tile
    {
      int n0 = wave * 16;
      v8f acc = {};
#pragma unroll 4
      for (int kt = 0; kt < Hd / 32; ++kt) {
        v16bf a = load_A_packed(sh_hA, kt, lane);
        v16bf b = load_B(Wq_bf, Hd, n0, kt * 32, lane);
        acc = WMMA_BF16(a, b, acc);
      }
      int n = n0 + (lane & 15);
      int mb = (lane >> 4) << 3;
      float bv = bq[n];
#pragma unroll
      for (int v = 0; v < 8; ++v) sh_pq[(mb + v) * Ed + n] = acc[v] + bv;
    }
    __syncthreads();

    // ---- scores[b][l] = M[b,l,:]·pq[b,:] with mask (f32 VALU, 4 per thread)
    for (int i = tid; i < 16 * Lq; i += 512) {
      int r = i >> 7, l = i & 127;
      const float4* Mp = (const float4*)(M + ((size_t)(b0 + r) * Lq + l) * Ed);
      const float4* q  = (const float4*)(sh_pq + r * Ed);
      float s = 0.0f;
      for (int e = 0; e < Ed / 4; ++e) {
        float4 m4 = Mp[e], q4 = q[e];
        s += m4.x * q4.x + m4.y * q4.y + m4.z * q4.z + m4.w * q4.w;
      }
      sh_w[i] = mask[(b0 + r) * Lq + l] ? -__builtin_inff() : s;
    }
    __syncthreads();

    // ---- softmax over L (wave = one row, wave32 shuffle reductions)
    {
      int r = wave;
      float mx = -__builtin_inff();
      for (int l = lane; l < Lq; l += 32) mx = fmaxf(mx, sh_w[r * Lq + l]);
      for (int off = 16; off; off >>= 1) mx = fmaxf(mx, __shfl_xor(mx, off, 32));
      float sum = 0.0f;
      for (int l = lane; l < Lq; l += 32) {
        float ev = __expf(sh_w[r * Lq + l] - mx);
        sh_w[r * Lq + l] = ev; sum += ev;
      }
      for (int off = 16; off; off >>= 1) sum += __shfl_xor(sum, off, 32);
      float inv = 1.0f / sum;
      for (int l = lane; l < Lq; l += 32) sh_w[r * Lq + l] *= inv;
      // emit this row's attention weights at its clipped length
      if (row_len == t)
        for (int l = lane; l < Lq; l += 32)
          out[(size_t)(b0 + r) * Lq + l] = sh_w[r * Lq + l];
    }

    // ---- applied[b,:] = sum_l w[l]*M[b,l,:] (wave = own row; 8 cols/lane),
    //      written straight into bf16 A-packed layout
    {
      int r = wave, e0 = lane * 8;
      float av[8];
#pragma unroll
      for (int j = 0; j < 8; ++j) av[j] = 0.0f;
      const float* Mb = M + ((size_t)(b0 + r) * Lq) * Ed + e0;
      for (int l = 0; l < Lq; ++l) {
        float wv = sh_w[r * Lq + l];
        const float4* mp = (const float4*)(Mb + (size_t)l * Ed);
        float4 a0 = mp[0], a1 = mp[1];
        av[0] += wv * a0.x; av[1] += wv * a0.y; av[2] += wv * a0.z; av[3] += wv * a0.w;
        av[4] += wv * a1.x; av[5] += wv * a1.y; av[6] += wv * a1.z; av[7] += wv * a1.w;
      }
      v8bf pk;
#pragma unroll
      for (int j = 0; j < 8; ++j) pk[j] = f2bf(av[j]);
      *(v8bf*)a_ptr8(sh_appA, r, e0) = pk;
    }

    // ---- stage x = emb[msg[:,t]] into bf16 A-packed layout (wave = own row)
    if (t < Td) {
      int r = wave, e0 = lane * 8;
      const float* xr = emb + (size_t)msg[(b0 + r) * Td + t] * Ed + e0;
      float4 x0 = ((const float4*)xr)[0], x1 = ((const float4*)xr)[1];
      v8bf pk;
      pk[0] = f2bf(x0.x); pk[1] = f2bf(x0.y); pk[2] = f2bf(x0.z); pk[3] = f2bf(x0.w);
      pk[4] = f2bf(x1.x); pk[5] = f2bf(x1.y); pk[6] = f2bf(x1.z); pk[7] = f2bf(x1.w);
      *(v8bf*)a_ptr8(sh_xA, r, e0) = pk;
    }
    __syncthreads();

    if (t == Td) break;  // final attend done; no LSTM at t==T

    // ---- GEMM2: att = [applied | h] @ Wc^T + bc  (16x512, K=768); 2 tiles/wave
    for (int j = 0; j < 2; ++j) {
      int n0 = (wave * 2 + j) * 16;
      v8f acc = {};
#pragma unroll 4
      for (int kt = 0; kt < Ed / 32; ++kt) {
        v16bf a = load_A_packed(sh_appA, kt, lane);
        v16bf b = load_B(Wc_bf, Ed + Hd, n0, kt * 32, lane);
        acc = WMMA_BF16(a, b, acc);
      }
#pragma unroll 4
      for (int kt = 0; kt < Hd / 32; ++kt) {
        v16bf a = load_A_packed(sh_hA, kt, lane);
        v16bf b = load_B(Wc_bf, Ed + Hd, n0, Ed + kt * 32, lane);
        acc = WMMA_BF16(a, b, acc);
      }
      int n = n0 + (lane & 15);
      int mb = (lane >> 4) << 3;
      float bv = bc[n];
#pragma unroll
      for (int v = 0; v < 8; ++v) a_store(sh_attA, mb + v, n, acc[v] + bv);
    }
    __syncthreads();

    // ---- GEMM3: gates = x@W_ih^T + att@W_hh^T, fused LSTM nonlinearity.
    for (int j = 0; j < 2; ++j) {
      int n0 = (wave * 2 + j) * 16;   // column tile within H; 4 gates together
      v8f ai = {}, af_ = {}, ag = {}, ao = {};
#pragma unroll 2
      for (int kt = 0; kt < Ed / 32; ++kt) {
        v16bf a = load_A_packed(sh_xA, kt, lane);
        int k0 = kt * 32;
        ai  = WMMA_BF16(a, load_B(Wih_bf, Ed, 0 * Hd + n0, k0, lane), ai);
        af_ = WMMA_BF16(a, load_B(Wih_bf, Ed, 1 * Hd + n0, k0, lane), af_);
        ag  = WMMA_BF16(a, load_B(Wih_bf, Ed, 2 * Hd + n0, k0, lane), ag);
        ao  = WMMA_BF16(a, load_B(Wih_bf, Ed, 3 * Hd + n0, k0, lane), ao);
      }
#pragma unroll 2
      for (int kt = 0; kt < Hd / 32; ++kt) {
        v16bf a = load_A_packed(sh_attA, kt, lane);
        int k0 = kt * 32;
        ai  = WMMA_BF16(a, load_B(Whh_bf, Hd, 0 * Hd + n0, k0, lane), ai);
        af_ = WMMA_BF16(a, load_B(Whh_bf, Hd, 1 * Hd + n0, k0, lane), af_);
        ag  = WMMA_BF16(a, load_B(Whh_bf, Hd, 2 * Hd + n0, k0, lane), ag);
        ao  = WMMA_BF16(a, load_B(Whh_bf, Hd, 3 * Hd + n0, k0, lane), ao);
      }
      int n  = n0 + (lane & 15);
      int mb = (lane >> 4) << 3;
      float bi  = b_ih[0 * Hd + n] + b_hh[0 * Hd + n];
      float bf_ = b_ih[1 * Hd + n] + b_hh[1 * Hd + n];
      float bg  = b_ih[2 * Hd + n] + b_hh[2 * Hd + n];
      float bo  = b_ih[3 * Hd + n] + b_hh[3 * Hd + n];
#pragma unroll
      for (int v = 0; v < 8; ++v) {
        int m = mb + v;                       // each (m,n) owned by one thread
        float cold = sh_c[m * Hd + n];
        float cn = sigf(af_[v] + bf_) * cold + sigf(ai[v] + bi) * tanhf(ag[v] + bg);
        float hn = sigf(ao[v] + bo) * tanhf(cn);
        sh_c[m * Hd + n] = cn;
        a_store(sh_hA, m, n, hn);             // h kept only as packed bf16
      }
    }
    __syncthreads();
  }
}

extern "C" void kernel_launch(void* const* d_in, const int* in_sizes, int n_in,
                              void* d_out, int out_size, void* d_ws, size_t ws_size,
                              hipStream_t stream) {
  const int*   msg  = (const int*)d_in[0];
  const float* M    = (const float*)d_in[1];
  const int*   mask = (const int*)d_in[2];
  const int*   len  = (const int*)d_in[3];
  const float* emb  = (const float*)d_in[4];
  const float* Wq   = (const float*)d_in[5];
  const float* bq   = (const float*)d_in[6];
  const float* Wc   = (const float*)d_in[7];
  const float* bc   = (const float*)d_in[8];
  const float* Wih  = (const float*)d_in[9];
  const float* Whh  = (const float*)d_in[10];
  const float* bih  = (const float*)d_in[11];
  const float* bhh  = (const float*)d_in[12];
  float* out = (float*)d_out;

  // bf16 weight staging in workspace (4 MB total, all regions 32B-aligned)
  __bf16* Wq_bf  = (__bf16*)d_ws;
  __bf16* Wc_bf  = Wq_bf  + Ed * Hd;
  __bf16* Wih_bf = Wc_bf  + Hd * (Ed + Hd);
  __bf16* Whh_bf = Wih_bf + 4 * Hd * Ed;

  convert_f32_to_bf16<<<(Ed * Hd + 255) / 256, 256, 0, stream>>>(Wq, Wq_bf, Ed * Hd);
  convert_f32_to_bf16<<<(Hd * (Ed + Hd) + 255) / 256, 256, 0, stream>>>(Wc, Wc_bf, Hd * (Ed + Hd));
  convert_f32_to_bf16<<<(4 * Hd * Ed + 255) / 256, 256, 0, stream>>>(Wih, Wih_bf, 4 * Hd * Ed);
  convert_f32_to_bf16<<<(4 * Hd * Hd + 255) / 256, 256, 0, stream>>>(Whh, Whh_bf, 4 * Hd * Hd);

  // logits/entropy outputs are zeros
  zero_tail<<<(2 * BSZ + 255) / 256, 256, 0, stream>>>(out, BSZ * Lq, 2 * BSZ);

  size_t smem = (size_t)(16 * Hd + 16 * Ed + 16 * Lq) * sizeof(float)
              + (size_t)(2 * 16 * Hd + 2 * 16 * Ed) * sizeof(__bf16);   // 104KB
  attn_lstm_kernel<<<BSZ / 16, 512, smem, stream>>>(msg, M, mask, len, emb, bq, bc,
                                                    bih, bhh, Wq_bf, Wc_bf, Wih_bf,
                                                    Whh_bf, out);
}
